// Qwen3_5Experts_11802570129921
// MI455X (gfx1250) — compile-verified, matrix operand order checked by reading, not run on hardware
//
#include <hip/hip_runtime.h>
#include <hip/hip_bf16.h>
#include <cstddef>

// Problem constants (match the reference)
constexpr int T_TOK = 1024;
constexpr int KTOP  = 8;
constexpr int NEXP  = 64;
constexpr int HDIM  = 1024;
constexpr int IDIM  = 512;
constexpr int CAP   = 4 * (T_TOK * KTOP) / NEXP; // 512
constexpr int MT    = 32;                        // M-tile rows per block

typedef __attribute__((ext_vector_type(16))) __bf16 bf16x16;
typedef __attribute__((ext_vector_type(8)))  __bf16 bf16x8;
typedef __attribute__((ext_vector_type(8)))  float  floatx8;

union Frag16 { bf16x16 v; bf16x8 h[2]; };

// ---------------------------------------------------------------------------
// A-fragment (16x32 bf16, M x K) from LDS. ISA layout: lanes 0-15 hold row M
// with K = [kk..kk+7] / [kk+16..kk+23]; lanes 16-31 hold K = [kk+8..kk+15] /
// [kk+24..kk+31]. Two contiguous 16B ds_load_b128 per lane.
// ---------------------------------------------------------------------------
__device__ inline bf16x16 load_a_lds(const __bf16* base, int rowStride, int kk, int lane) {
    const int r  = lane & 15;
    const int kh = (lane >> 4) * 8;
    const __bf16* row = base + r * rowStride;
    Frag16 f;
    f.h[0] = *(const bf16x8*)(row + kk + kh);
    f.h[1] = *(const bf16x8*)(row + kk + 16 + kh);
    return f.v;
}

// ---------------------------------------------------------------------------
// B-fragment (32x16 bf16, K x N) from row-major fp32 weights whose rows are
// the N (output) dimension: B[k][n] = W[(n0+n)*rowStride + kk+k].
// 4x global_load_b128 + 8x v_cvt_pk_bf16_f32 per lane; prefetch next k-step.
// ---------------------------------------------------------------------------
__device__ inline bf16x16 load_b_global(const float* __restrict__ W, int rowStride,
                                        int n0, int kk, int lane) {
    const int n  = lane & 15;
    const int kh = (lane >> 4) * 8;
    const float* row = W + (size_t)(n0 + n) * rowStride;
    const float4 f0 = *(const float4*)(row + kk + kh);
    const float4 f1 = *(const float4*)(row + kk + kh + 4);
    const float4 f2 = *(const float4*)(row + kk + 16 + kh);
    const float4 f3 = *(const float4*)(row + kk + 16 + kh + 4);
    // speculative prefetch of the next k-step of this row (dropped if invalid)
    __builtin_prefetch(row + kk + 32 + kh, 0, 1);
    bf16x16 b;
    b[0]  = (__bf16)f0.x; b[1]  = (__bf16)f0.y; b[2]  = (__bf16)f0.z; b[3]  = (__bf16)f0.w;
    b[4]  = (__bf16)f1.x; b[5]  = (__bf16)f1.y; b[6]  = (__bf16)f1.z; b[7]  = (__bf16)f1.w;
    b[8]  = (__bf16)f2.x; b[9]  = (__bf16)f2.y; b[10] = (__bf16)f2.z; b[11] = (__bf16)f2.w;
    b[12] = (__bf16)f3.x; b[13] = (__bf16)f3.y; b[14] = (__bf16)f3.z; b[15] = (__bf16)f3.w;
    return b;
}

__device__ inline floatx8 wmma_bf16(bf16x16 a, bf16x16 b, floatx8 c) {
    return __builtin_amdgcn_wmma_f32_16x16x32_bf16(
        /*neg_a=*/false, a, /*neg_b=*/false, b,
        /*c_mod=*/(short)0, c, /*reuse_a=*/false, /*reuse_b=*/false);
}

// ---------------------------------------------------------------------------
// Kernel 0: zero output, zero per-expert counts, invalidate slot table.
// ---------------------------------------------------------------------------
__global__ void init_kernel(float* __restrict__ out, int* __restrict__ counts,
                            int* __restrict__ pair_of_slot) {
    int i = blockIdx.x * blockDim.x + threadIdx.x;
    if (i < T_TOK * HDIM)  out[i] = 0.0f;
    if (i < NEXP)          counts[i] = 0;
    if (i < NEXP * CAP)    pair_of_slot[i] = -1;
}

// ---------------------------------------------------------------------------
// Kernel 1: routing via per-expert atomic counter. Scatter and gather use the
// same slot table, so the final sum is invariant to assignment order.
// ---------------------------------------------------------------------------
__global__ void route_kernel(const int* __restrict__ topk_idx,
                             int* __restrict__ counts,
                             int* __restrict__ pair_of_slot) {
    int p = blockIdx.x * blockDim.x + threadIdx.x;
    if (p >= T_TOK * KTOP) return;
    int e = topk_idx[p];
    int pos = atomicAdd(&counts[e], 1);
    if (pos < CAP) pair_of_slot[e * CAP + pos] = p;
}

// ---------------------------------------------------------------------------
// Kernel 2: fused per-expert GEMM1 (gate/up + SiLU) and GEMM2 (down) with
// weighted scatter-add. One block = one expert x one 32-row tile; 512 threads
// = 16 wave32 waves. Every B fragment is reused by 2 row sub-tiles.
// ---------------------------------------------------------------------------
__global__ __launch_bounds__(512)
void moe_expert_kernel(const float* __restrict__ x,
                       const float* __restrict__ topk_w,
                       const float* __restrict__ gup,
                       const float* __restrict__ down,
                       const int*   __restrict__ counts,
                       const int*   __restrict__ pair_of_slot,
                       float* __restrict__ out) {
    const int e       = blockIdx.x;
    const int rowTile = blockIdx.y;

    int cnt = counts[e];
    if (cnt > CAP) cnt = CAP;
    if (rowTile * MT >= cnt) return;   // ragged early-exit

    __shared__ __bf16 xs[MT * HDIM];   // gathered activations (bf16, 64 KB)
    __shared__ __bf16 hs[MT * IDIM];   // silu(gate)*up        (bf16, 32 KB)
    __shared__ int    ps[MT];          // pair index per tile row (-1 = padding)
    __shared__ float  wt[MT];          // routing weight per tile row

    const int tid  = threadIdx.x;
    const int lane = tid & 31;
    const int wave = tid >> 5;

    if (tid < MT) {
        int s = rowTile * MT + tid;
        int pair = (s < cnt) ? pair_of_slot[e * CAP + s] : -1;
        ps[tid] = pair;
        wt[tid] = (pair >= 0) ? topk_w[pair] : 0.0f;
    }
    __syncthreads();

    // Gather 32 token rows (fp32 -> bf16) into LDS: 16 threads/row, 16 float4 each.
    {
        const int r   = tid >> 4;       // 0..31
        const int c16 = tid & 15;
        const int pair = ps[r];
        const float* src = (pair >= 0) ? (x + (size_t)(pair / KTOP) * HDIM) : nullptr;
        for (int chunk = 0; chunk < 16; ++chunk) {
            const int col = (c16 + chunk * 16) * 4;
            float4 f = src ? *(const float4*)(src + col) : make_float4(0.f, 0.f, 0.f, 0.f);
            __bf16* d = xs + r * HDIM + col;
            d[0] = (__bf16)f.x; d[1] = (__bf16)f.y;
            d[2] = (__bf16)f.z; d[3] = (__bf16)f.w;
        }
    }
    __syncthreads();

    // ---- GEMM1: gu = x(32xH) * Wgu^T, fused SiLU-gate, h -> LDS -------------
    // Wave handles I-col tiles {wave, wave+16}, gate+up each, for both 16-row
    // sub-tiles: 8 accumulators, 8 wmma per 2 A-loads + 4 B-loads per k-step.
    const float* Wg = gup + (size_t)e * (2 * IDIM) * HDIM;
    {
        floatx8 accG[2][2] = {};   // [col-tile jj][row sub-tile]
        floatx8 accU[2][2] = {};
        for (int kk = 0; kk < HDIM; kk += 32) {
            bf16x16 a0 = load_a_lds(xs,             HDIM, kk, lane);
            bf16x16 a1 = load_a_lds(xs + 16 * HDIM, HDIM, kk, lane);
#pragma unroll
            for (int jj = 0; jj < 2; ++jj) {
                const int j = wave + jj * 16;          // 0..31
                bf16x16 bg = load_b_global(Wg, HDIM, j * 16,        kk, lane);
                accG[jj][0] = wmma_bf16(a0, bg, accG[jj][0]);
                accG[jj][1] = wmma_bf16(a1, bg, accG[jj][1]);
                bf16x16 bu = load_b_global(Wg, HDIM, IDIM + j * 16, kk, lane);
                accU[jj][0] = wmma_bf16(a0, bu, accU[jj][0]);
                accU[jj][1] = wmma_bf16(a1, bu, accU[jj][1]);
            }
        }
        const int n = lane & 15;
#pragma unroll
        for (int jj = 0; jj < 2; ++jj) {
            const int j = wave + jj * 16;
#pragma unroll
            for (int rt = 0; rt < 2; ++rt) {
                const int mbase = rt * 16 + (lane >> 4) * 8;
#pragma unroll
                for (int v = 0; v < 8; ++v) {
                    float g = accG[jj][rt][v];
                    float u = accU[jj][rt][v];
                    // silu(g) * u with a single v_rcp_f32 (no IEEE div expansion)
                    float s = __builtin_amdgcn_rcpf(1.0f + __expf(-g));
                    hs[(mbase + v) * IDIM + j * 16 + n] = (__bf16)(u * g * s);
                }
            }
        }
    }
    __syncthreads();

    // ---- GEMM2: y = h(32xI) * Wd^T, weighted scatter-add to tokens ----------
    // Wave handles H-col tiles {wave, wave+16, wave+32, wave+48} x 2 row tiles.
    const float* Wd = down + (size_t)e * HDIM * IDIM;
    {
        floatx8 acc[4][2] = {};
        for (int kk = 0; kk < IDIM; kk += 32) {
            bf16x16 a0 = load_a_lds(hs,             IDIM, kk, lane);
            bf16x16 a1 = load_a_lds(hs + 16 * IDIM, IDIM, kk, lane);
#pragma unroll
            for (int jj = 0; jj < 4; ++jj) {
                const int j = wave + jj * 16;          // 0..63
                bf16x16 b = load_b_global(Wd, IDIM, j * 16, kk, lane);
                acc[jj][0] = wmma_bf16(a0, b, acc[jj][0]);
                acc[jj][1] = wmma_bf16(a1, b, acc[jj][1]);
            }
        }
        const int n = lane & 15;
#pragma unroll
        for (int jj = 0; jj < 4; ++jj) {
            const int col = (wave + jj * 16) * 16 + n;
#pragma unroll
            for (int rt = 0; rt < 2; ++rt) {
                const int mbase = rt * 16 + (lane >> 4) * 8;
#pragma unroll
                for (int v = 0; v < 8; ++v) {
                    const int m = mbase + v;
                    const int pair = ps[m];
                    if (pair >= 0) {
                        const int t = pair / KTOP;
                        atomicAdd(out + (size_t)t * HDIM + col, acc[jj][rt][v] * wt[m]);
                    }
                }
            }
        }
    }
}

// ---------------------------------------------------------------------------
// Launch wrapper
// ---------------------------------------------------------------------------
extern "C" void kernel_launch(void* const* d_in, const int* in_sizes, int n_in,
                              void* d_out, int out_size, void* d_ws, size_t ws_size,
                              hipStream_t stream) {
    const float* x      = (const float*)d_in[0];   // (T, H)
    const int*   idx    = (const int*)  d_in[1];   // (T, K)
    const float* w      = (const float*)d_in[2];   // (T, K)
    const float* gup    = (const float*)d_in[3];   // (E, 2I, H)
    const float* down   = (const float*)d_in[4];   // (E, H, I)
    float*       out    = (float*)d_out;           // (T, H)

    int* counts       = (int*)d_ws;                // E ints
    int* pair_of_slot = counts + NEXP;             // E*CAP ints (~128 KB)

    {
        int n = T_TOK * HDIM;
        init_kernel<<<(n + 255) / 256, 256, 0, stream>>>(out, counts, pair_of_slot);
    }
    {
        int n = T_TOK * KTOP;
        route_kernel<<<(n + 255) / 256, 256, 0, stream>>>(idx, counts, pair_of_slot);
    }
    {
        dim3 grid(NEXP, CAP / MT);
        moe_expert_kernel<<<grid, 512, 0, stream>>>(x, w, gup, down,
                                                    counts, pair_of_slot, out);
    }
}
